// CapInfoNCE_43619687859123
// MI455X (gfx1250) — matrix operand order repaired and checked by reading, not compiled, not run on hardware
//
#include <hip/hip_runtime.h>
#include <stdint.h>

typedef __attribute__((ext_vector_type(16))) __bf16 v16bf;
typedef __attribute__((ext_vector_type(8)))  __bf16 v8bf;
typedef __attribute__((ext_vector_type(8)))  float  v8f;
typedef __attribute__((ext_vector_type(4)))  unsigned int v4u;
typedef __attribute__((ext_vector_type(8)))  int v8i;
typedef __attribute__((ext_vector_type(4)))  int v4i;

#define B_   64
#define TO   36
#define TOP  48   // To padded to multiple of 16 (softmax-masked)
#define KP   64   // K padding for att@o GEMM (2 x K=32 WMMA steps)
#define TW   32
#define D_   512

// ---- workspace layout (element offsets in __bf16 units) ----
static constexpr size_t N_W   = (size_t)B_ * TW  * D_;   // 1,048,576
static constexpr size_t N_U   = (size_t)B_ * TOP * D_;   // 1,572,864
static constexpr size_t N_OT  = (size_t)B_ * D_  * KP;   // 2,097,152
static constexpr size_t OFF_W  = 0;
static constexpr size_t OFF_U  = OFF_W + N_W;
static constexpr size_t OFF_OT = OFF_U + N_U;
static constexpr size_t LOGITS_BYTE_OFF = (OFF_OT + N_OT) * 2;  // 9,437,184 (16B aligned)

// ------------------- f32 -> bf16 staging kernels -------------------
__global__ void conv_w_kernel(const float* __restrict__ w, __bf16* __restrict__ out) {
    size_t i = (size_t)blockIdx.x * blockDim.x + threadIdx.x;
    if (i < N_W) out[i] = (__bf16)w[i];
}

__global__ void conv_u_kernel(const float* __restrict__ u, __bf16* __restrict__ out) {
    size_t i = (size_t)blockIdx.x * blockDim.x + threadIdx.x;
    if (i >= N_U) return;
    int d = (int)(i % D_);
    size_t t = i / D_;
    int r = (int)(t % TOP);
    int b = (int)(t / TOP);
    float v = (r < TO) ? u[((size_t)b * TO + r) * D_ + d] : 0.0f;
    out[i] = (__bf16)v;
}

// o stored transposed per image: out[b][d][k] (k padded to KP with zeros)
__global__ void conv_oT_kernel(const float* __restrict__ o, __bf16* __restrict__ out) {
    size_t i = (size_t)blockIdx.x * blockDim.x + threadIdx.x;
    if (i >= N_OT) return;
    int k = (int)(i % KP);
    size_t t = i / KP;
    int d = (int)(t % D_);
    int b = (int)(t / D_);
    float v = (k < TO) ? o[((size_t)b * TO + k) * D_ + d] : 0.0f;
    out[i] = (__bf16)v;
}

// ------------------- TDM: async tensor load (2D bf16 tile, contiguous rows) -------------------
// D# group0: [1:0]=count=1 | [63:32]=lds_addr | [120:64]=global_addr | [127:126]=type(2)
// D# group1: [17:16]=data_size(1 => 2B) | tensor_dim0/1, tile_dim0/1, tensor_dim0_stride
__device__ __forceinline__ void tdm_load_2d(const void* gptr, void* lptr,
                                            unsigned dim0, unsigned dim1,
                                            unsigned tile0, unsigned tile1,
                                            unsigned stride0)
{
    unsigned long long ga = (unsigned long long)(uintptr_t)gptr;
    unsigned la = (unsigned)(uintptr_t)lptr;     // LDS byte offset (addr[31:0])
    v4u g0;
    g0.x = 1u;                                   // count=1, user descriptor
    g0.y = la;                                   // lds_addr
    g0.z = (unsigned)(ga & 0xFFFFFFFFu);         // global_addr[31:0]
    g0.w = (unsigned)((ga >> 32) & 0x01FFFFFFu)  // global_addr[56:32]
         | (2u << 30);                           // type = 2 ("image")
    v8i g1;
    g1[0] = (int)(1u << 16);                     // wg_mask=0, data_size=1 (2 bytes)
    g1[1] = (int)((dim0 & 0xFFFFu) << 16);       // tensor_dim0[15:0] in dword1[31:16]
    g1[2] = (int)((dim0 >> 16) | ((dim1 & 0xFFFFu) << 16));  // dim0 hi | dim1 lo
    g1[3] = (int)((dim1 >> 16) | (tile0 << 16)); // dim1 hi | tile_dim0
    g1[4] = (int)(tile1);                        // tile_dim1 | tile_dim2=0
    g1[5] = (int)(stride0);                      // tensor_dim0_stride[31:0]
    g1[6] = 0;                                   // stride0[47:32]=0 | stride1 lo=0
    g1[7] = 0;                                   // stride1 hi=0
    v4i z4 = {0, 0, 0, 0};                       // groups 2/3 unused (2D tensor)
    v8i z8 = {0, 0, 0, 0, 0, 0, 0, 0};
    __builtin_amdgcn_tensor_load_to_lds(g0, g1, z4, z4, z8, 0);
}

// ------------------- main fused attention kernel -------------------
union ABOp { v16bf v; v8bf h[2]; };

__launch_bounds__(128)
__global__ void capnce_main(const __bf16* __restrict__ gw,
                            const __bf16* __restrict__ gu,
                            const __bf16* __restrict__ goT,
                            float* __restrict__ att_out,
                            float* __restrict__ attv_out,
                            float* __restrict__ logits_out)
{
    __shared__ __bf16 lds_w [TW  * D_];   // 32 KB
    __shared__ __bf16 lds_u [TOP * D_];   // 48 KB
    __shared__ __bf16 lds_oT[D_  * KP];   // 64 KB
    __shared__ float  lds_s [TW  * TOP];  // 6 KB
    __shared__ __bf16 lds_att[TW * KP];   // 4 KB
    __shared__ float  lds_logit[TW];

    const int tid  = threadIdx.x;
    const int wave = tid >> 5;
    const int lane = tid & 31;
    const int wb = blockIdx.x, ob = blockIdx.y;

    if (tid < TW) lds_logit[tid] = 0.0f;

    // ---- stage tiles global -> LDS via Tensor Data Mover (wave 0 issues DMAs) ----
    if (wave == 0) {
        tdm_load_2d(gw  + (size_t)wb * TW  * D_, lds_w,  D_, TW,  D_, TW,  D_);
        tdm_load_2d(gu  + (size_t)ob * TOP * D_, lds_u,  D_, TOP, D_, TOP, D_);
        tdm_load_2d(goT + (size_t)ob * D_  * KP, lds_oT, KP, D_,  KP, D_,  KP);
        __builtin_amdgcn_s_wait_tensorcnt(0);    // TENSORcnt == 0
    }
    __syncthreads();

    // ---- GEMM 1: scores(32x48) = w(32x512) @ u^T, K-steps of 32 (bf16 WMMA) ----
    const float scale = 0.04419417382415922f;   // 1/sqrt(512)
    const int kh = lane >> 4;                   // lane-half selects K sub-pattern
    for (int tile = wave; tile < 6; tile += 4) {
        const int mt = tile & 1, nt = tile >> 1;
        const int m = mt * 16 + (lane & 15);    // A row (caption token)
        const int n = nt * 16 + (lane & 15);    // B column (object)
        v8f acc = {0.f,0.f,0.f,0.f,0.f,0.f,0.f,0.f};
        #pragma unroll 4
        for (int k0 = 0; k0 < D_; k0 += 32) {
            ABOp a, b;
            a.h[0] = *(const v8bf*)&lds_w[m * D_ + k0 +      kh * 8];
            a.h[1] = *(const v8bf*)&lds_w[m * D_ + k0 + 16 + kh * 8];
            b.h[0] = *(const v8bf*)&lds_u[n * D_ + k0 +      kh * 8];
            b.h[1] = *(const v8bf*)&lds_u[n * D_ + k0 + 16 + kh * 8];
            acc = __builtin_amdgcn_wmma_f32_16x16x32_bf16(
                false, a.v, false, b.v, (short)0, acc, false, false);
        }
        #pragma unroll
        for (int r = 0; r < 8; ++r) {
            int t = mt * 16 + r + kh * 8;       // C/D layout: lanes 16-31 -> M+8
            float v = acc[r] * scale;
            lds_s[t * TOP + n] = (n < TO) ? v : -1e30f;  // mask padded objects
        }
    }
    __syncthreads();

    // ---- softmax over objects (per caption-token row) ----
    if (tid < TW) {
        const int t = tid;
        float mx = -1e30f;
        for (int c = 0; c < TOP; ++c) mx = fmaxf(mx, lds_s[t * TOP + c]);
        float sum = 0.0f;
        for (int c = 0; c < TOP; ++c) {
            float e = __expf(lds_s[t * TOP + c] - mx);
            lds_s[t * TOP + c] = e;
            sum += e;
        }
        float inv = 1.0f / sum;
        for (int c = 0; c < TOP; ++c) lds_s[t * TOP + c] *= inv;
    } else if (tid < 64) {
        const int t = tid - 32;                 // zero K-padding columns 48..63
        for (int c = TOP; c < KP; ++c) lds_att[t * KP + c] = (__bf16)0.0f;
    }
    __syncthreads();

    // ---- write att output (coalesced) + build bf16 att operand ----
    {
        const size_t base = ((size_t)(wb * B_ + ob)) * TW * TO;
        for (int i = tid; i < TW * TO; i += 128) {
            int t = i / TO, c = i % TO;
            att_out[base + i] = lds_s[t * TOP + c];
        }
        for (int i = tid; i < TW * TOP; i += 128) {
            int t = i / TOP, c = i % TOP;
            lds_att[t * KP + c] = (__bf16)lds_s[i];
        }
    }
    __syncthreads();

    // ---- GEMM 2: att_V(32x512) = att(32x64) @ o(64x512) + fused logits ----
    {
        const int mt = wave & 1;                // fixed parity per wave
        const int m  = mt * 16 + (lane & 15);
        const size_t avbase = ((size_t)(wb * B_ + ob)) * TW * D_;

        ABOp a0, a1;                            // A operands reused for all 16 n-tiles
        a0.h[0] = *(const v8bf*)&lds_att[m * KP +  0 + kh * 8];
        a0.h[1] = *(const v8bf*)&lds_att[m * KP + 16 + kh * 8];
        a1.h[0] = *(const v8bf*)&lds_att[m * KP + 32 + kh * 8];
        a1.h[1] = *(const v8bf*)&lds_att[m * KP + 48 + kh * 8];

        float part[8] = {0.f,0.f,0.f,0.f,0.f,0.f,0.f,0.f};
        for (int tile = wave; tile < 64; tile += 4) {
            const int nt2  = tile >> 1;
            const int dcol = nt2 * 16 + (lane & 15);
            v8f acc = {0.f,0.f,0.f,0.f,0.f,0.f,0.f,0.f};
            ABOp b;
            b.h[0] = *(const v8bf*)&lds_oT[dcol * KP +  0 + kh * 8];
            b.h[1] = *(const v8bf*)&lds_oT[dcol * KP + 16 + kh * 8];
            acc = __builtin_amdgcn_wmma_f32_16x16x32_bf16(
                false, a0.v, false, b.v, (short)0, acc, false, false);
            b.h[0] = *(const v8bf*)&lds_oT[dcol * KP + 32 + kh * 8];
            b.h[1] = *(const v8bf*)&lds_oT[dcol * KP + 48 + kh * 8];
            acc = __builtin_amdgcn_wmma_f32_16x16x32_bf16(
                false, a1.v, false, b.v, (short)0, acc, false, false);
            #pragma unroll
            for (int r = 0; r < 8; ++r) {
                int t = mt * 16 + r + kh * 8;
                float v = acc[r];
                attv_out[avbase + (size_t)t * D_ + dcol] = v;       // stream out once
                part[r] += v * (float)lds_w[t * D_ + dcol];         // fused logits dot
            }
        }
        #pragma unroll
        for (int r = 0; r < 8; ++r) {
            int t = mt * 16 + r + kh * 8;
            atomicAdd(&lds_logit[t], part[r]);                      // ds_add_f32
        }
    }
    __syncthreads();
    if (tid < TW)
        logits_out[((size_t)(wb * B_ + ob)) * TW + tid] = lds_logit[tid];
}

// ------------------- loss kernel: log-softmax over ob + diag reduction -------------------
__global__ void capnce_loss(const float* __restrict__ logits,
                            const int* __restrict__ mask,
                            float* __restrict__ loss_out)
{
    const int wb = blockIdx.x;
    const int t  = threadIdx.x;                 // 32 threads = 1 wave
    float mx = -1e30f;
    for (int ob = 0; ob < B_; ++ob)
        mx = fmaxf(mx, logits[((size_t)(wb * B_ + ob)) * TW + t]);
    float s = 0.0f;
    for (int ob = 0; ob < B_; ++ob)
        s += __expf(logits[((size_t)(wb * B_ + ob)) * TW + t] - mx);
    const float lse  = mx + __logf(s);
    const float diag = logits[((size_t)(wb * B_ + wb)) * TW + t];
    float keep = (mask[wb * TW + t] != 0) ? 0.0f : 1.0f;
    float val  = (diag - lse) * keep;
    for (int off = 16; off > 0; off >>= 1) {
        val  += __shfl_down(val,  off, 32);
        keep += __shfl_down(keep, off, 32);
    }
    if (t == 0) {
        float contrib = val / (keep + 1e-6f);
        atomicAdd(loss_out, -contrib / (float)B_);
    }
}

// ------------------- launcher -------------------
extern "C" void kernel_launch(void* const* d_in, const int* in_sizes, int n_in,
                              void* d_out, int out_size, void* d_ws, size_t ws_size,
                              hipStream_t stream) {
    (void)in_sizes; (void)n_in; (void)out_size; (void)ws_size;
    const float* o    = (const float*)d_in[0];
    const float* u    = (const float*)d_in[1];
    const float* w    = (const float*)d_in[2];
    const int*   mask = (const int*)d_in[3];

    float* out  = (float*)d_out;
    float* loss = out;                                   // [1]
    float* att  = out + 1;                               // [64,64,32,36]
    float* attv = att + (size_t)B_ * B_ * TW * TO;       // [64,64,32,512]

    __bf16* ws  = (__bf16*)d_ws;
    __bf16* hw  = ws + OFF_W;
    __bf16* hu  = ws + OFF_U;
    __bf16* hoT = ws + OFF_OT;
    float* logits = (float*)((char*)d_ws + LOGITS_BYTE_OFF);  // [64,64,32]

    (void)hipMemsetAsync(d_out, 0, sizeof(float), stream);   // zero loss accumulator

    conv_w_kernel <<<(unsigned)((N_W  + 255) / 256), 256, 0, stream>>>(w, hw);
    conv_u_kernel <<<(unsigned)((N_U  + 255) / 256), 256, 0, stream>>>(u, hu);
    conv_oT_kernel<<<(unsigned)((N_OT + 255) / 256), 256, 0, stream>>>(o, hoT);

    capnce_main<<<dim3(B_, B_), 128, 0, stream>>>(hw, hu, hoT, att, attv, logits);
    capnce_loss<<<B_, 32, 0, stream>>>(logits, mask, loss);
}
